// FlashNeoxAttention_74440373174904
// MI455X (gfx1250) — matrix operand order repaired and checked by reading, not compiled, not run on hardware
//
#include <hip/hip_runtime.h>
#include <hip/hip_bf16.h>

// ---------------- problem constants (from reference) ----------------
constexpr int T_    = 4096;   // total tokens
constexpr int HID_  = 2048;   // hidden size
constexpr int NH_   = 16;     // heads
constexpr int HD_   = 128;    // head dim
constexpr int SEQ_  = 2048;   // seq len
constexpr int BATCH_= 2;
constexpr int NSLOTS_ = 16384;

// ---------------- CDNA5 WMMA / TDM types ----------------
typedef __attribute__((ext_vector_type(16))) __bf16 v16bf;
typedef __attribute__((ext_vector_type(8)))  float  v8f;
typedef __attribute__((ext_vector_type(4)))  unsigned int u32x4;
typedef __attribute__((ext_vector_type(4)))  int  i32x4;
typedef __attribute__((ext_vector_type(8)))  int  i32x8;

#define WMMA_BF16(a, b, c) \
  __builtin_amdgcn_wmma_f32_16x16x32_bf16(false, (a), false, (b), (short)0, (c), false, false)

__device__ __forceinline__ unsigned short f2bf(float f) {
  union { float f; unsigned u; } v; v.f = f;
  unsigned r = v.u + 0x7FFFu + ((v.u >> 16) & 1u);   // round-to-nearest-even
  return (unsigned short)(r >> 16);
}
__device__ __forceinline__ __bf16 bfbits(unsigned short u) {
  union { unsigned short u; __bf16 b; } c; c.u = u; return c.b;
}
__device__ __forceinline__ __bf16 f2bf16(float f) { return bfbits(f2bf(f)); }

// ---------------------------------------------------------------------
// TDM: async 2D tile load, global -> LDS  (32 rows x 128 fp32,
// row stride NH*HD elements, LDS padded +2 DWORDs per 128-DWORD row).
// Descriptor packing per CDNA5 ISA.
// ---------------------------------------------------------------------
__device__ __forceinline__ void tdm_load_tile_32x128(unsigned lds_off,
                                                     const float* gsrc) {
  unsigned long long ga = (unsigned long long)(uintptr_t)gsrc;
  u32x4 g0;
  g0[0] = 1u;                                           // count=1, user desc
  g0[1] = lds_off;                                      // lds_addr (bytes)
  g0[2] = (unsigned)(ga & 0xFFFFFFFFull);               // global_addr[31:0]
  g0[3] = (unsigned)((ga >> 32) & 0x01FFFFFFull)        // global_addr[56:32]
        | (2u << 30);                                   // type=2 ("image")
  i32x8 g1;
  g1[0] = (2 << 16)        // data_size = 4B
        | (1 << 20)        // pad_enable
        | (6 << 22)        // pad_interval: 128 DWORDs
        | (1 << 25);       // pad_amount: 2 DWORDs  -> LDS row stride 130 f32
  g1[1] = (HD_ << 16);     // tensor_dim0[15:0] = 128
  g1[2] = (32 << 16);      // tensor_dim0[31:16]=0 | tensor_dim1[15:0] = 32
  g1[3] = (HD_ << 16);     // tensor_dim1[31:16]=0 | tile_dim0 = 128
  g1[4] = 32;              // tile_dim1 = 32, tile_dim2 = 0 (2D)
  g1[5] = NH_ * HD_;       // tensor_dim0_stride = 2048 elements
  g1[6] = 0;               // stride hi, tensor_dim1_stride lo
  g1[7] = 0;
  i32x4 z4 = {0, 0, 0, 0};
#if defined(__clang_major__) && (__clang_major__ >= 23)
  i32x8 z8 = {0, 0, 0, 0, 0, 0, 0, 0};
  __builtin_amdgcn_tensor_load_to_lds(g0, g1, z4, z4, z8, 0);
#else
  __builtin_amdgcn_tensor_load_to_lds(g0, g1, z4, z4, 0);
#endif
}

// =====================================================================
// Generic  C[M,N] = A[M,K] @ W[N,K]^T + bias[N]   (bf16 WMMA, f32 acc)
// block: 256 thr (8 waves), tile 64x64, k-step 32.
// =====================================================================
__global__ __launch_bounds__(256) void gemm_bias_wmma(
    const float* __restrict__ A, const float* __restrict__ W,
    const float* __restrict__ bias, float* __restrict__ C,
    int M, int N, int K) {
  __shared__ unsigned short sA[64 * 34];
  __shared__ unsigned short sW[64 * 34];
  const int tid  = threadIdx.x;
  const int lane = tid & 31, wid = tid >> 5;
  const int g = lane >> 4, ln = lane & 15;
  const int wm = wid & 3, wn = wid >> 2;
  const int m0 = blockIdx.y * 64, n0 = blockIdx.x * 64;
  const int row = tid >> 2;            // 0..63
  const int seg = (tid & 3) * 8;       // 0,8,16,24

  v8f acc0 = {}; v8f acc1 = {};

  for (int k0 = 0; k0 < K; k0 += 32) {
    // cooperative fp32->bf16 staging into LDS
    const float4* ap = (const float4*)(A + (size_t)(m0 + row) * K + k0 + seg);
    const float4* wp = (const float4*)(W + (size_t)(n0 + row) * K + k0 + seg);
    float4 a0 = ap[0], a1 = ap[1];
    float4 w0 = wp[0], w1 = wp[1];
    if (k0 + 32 < K) {   // gfx1250 global_prefetch_b8, WGP scope (locality 3)
      __builtin_prefetch(A + (size_t)(m0 + row) * K + k0 + 32 + seg, 0, 3);
      __builtin_prefetch(W + (size_t)(n0 + row) * K + k0 + 32 + seg, 0, 3);
    }
    unsigned short* da = &sA[row * 34 + seg];
    unsigned short* dw = &sW[row * 34 + seg];
    da[0]=f2bf(a0.x); da[1]=f2bf(a0.y); da[2]=f2bf(a0.z); da[3]=f2bf(a0.w);
    da[4]=f2bf(a1.x); da[5]=f2bf(a1.y); da[6]=f2bf(a1.z); da[7]=f2bf(a1.w);
    dw[0]=f2bf(w0.x); dw[1]=f2bf(w0.y); dw[2]=f2bf(w0.z); dw[3]=f2bf(w0.w);
    dw[4]=f2bf(w1.x); dw[5]=f2bf(w1.y); dw[6]=f2bf(w1.z); dw[7]=f2bf(w1.w);
    __syncthreads();

    // A fragment: lane ln = row m, K striped per ISA 16-bit A layout
    v16bf af;
#pragma unroll
    for (int j = 0; j < 16; ++j) {
      int k = 8 * g + (j & 7) + ((j >> 3) << 4);
      af[j] = bfbits(sA[(wm * 16 + ln) * 34 + k]);
    }
    // B fragments: B[k][n] = W[n][k]
    v16bf bf0, bf1;
#pragma unroll
    for (int j = 0; j < 16; ++j) {
      int k = 8 * g + (j & 7) + ((j >> 3) << 4);
      bf0[j] = bfbits(sW[(wn * 32 + ln) * 34 + k]);
      bf1[j] = bfbits(sW[(wn * 32 + 16 + ln) * 34 + k]);
    }
    acc0 = WMMA_BF16(af, bf0, acc0);
    acc1 = WMMA_BF16(af, bf1, acc1);
    __syncthreads();
  }

  // epilogue: D layout — vgpr r holds row r + 8*(lane>=16), col = lane&15
  const int nc0 = n0 + wn * 32 + ln;
  const int nc1 = nc0 + 16;
  const float b0 = bias[nc0], b1 = bias[nc1];
#pragma unroll
  for (int r = 0; r < 8; ++r) {
    int mr = m0 + wm * 16 + r + 8 * g;
    C[(size_t)mr * N + nc0] = acc0[r] + b0;
    C[(size_t)mr * N + nc1] = acc1[r] + b1;
  }
}

// =====================================================================
// RoPE (first 32 dims of each head) + KV-cache scatter.
// =====================================================================
__global__ __launch_bounds__(256) void rope_scatter(
    const float* __restrict__ qkv, const float* __restrict__ cosb,
    const float* __restrict__ sinb, const int* __restrict__ slots,
    float* __restrict__ qout, float* __restrict__ kc, float* __restrict__ vc) {
  int idx = blockIdx.x * blockDim.x + threadIdx.x;
  if (idx >= T_ * NH_ * HD_) return;
  int d = idx & (HD_ - 1);
  int h = (idx >> 7) & (NH_ - 1);
  int t = idx >> 11;
  const float* base = qkv + (size_t)t * 3 * HID_ + h * HD_;
  float qv = base[d];
  float kv = base[HID_ + d];
  float vv = base[2 * HID_ + d];
  if (d < 32) {
    if (d < 16) {
      float c = cosb[t * 16 + d], s = sinb[t * 16 + d];
      float q2 = base[d + 16], k2 = base[HID_ + d + 16];
      qv = qv * c - q2 * s;
      kv = kv * c - k2 * s;
    } else {
      float c = cosb[t * 16 + d - 16], s = sinb[t * 16 + d - 16];
      float q1 = base[d - 16], k1 = base[HID_ + d - 16];
      qv = qv * c + q1 * s;
      kv = kv * c + k1 * s;
    }
  }
  qout[(size_t)t * HID_ + h * HD_ + d] = qv;
  int slot = slots[t];
  size_t ci = ((size_t)slot * NH_ + h) * HD_ + d;
  kc[ci] = kv;
  vc[ci] = vv;
}

// =====================================================================
// Causal flash attention, bf16 WMMA + TDM-staged K/V tiles.
// block = 4 waves = 64 queries of one (batch, head). Key blocks of 32.
// =====================================================================
__global__ __launch_bounds__(128) void flash_attn_wmma(
    const float* __restrict__ Q, const float* __restrict__ Kc,
    const float* __restrict__ Vc, float* __restrict__ O) {
  constexpr int LKV = 130;                       // padded LDS row stride (f32)
  __shared__ float sKf[32 * LKV];
  __shared__ float sVf[32 * LKV];
  __shared__ unsigned short sP[4][16 * 34];

  const int tid  = threadIdx.x;
  const int lane = tid & 31, wid = tid >> 5;
  const int g = lane >> 4, ln = lane & 15;
  const int q0 = blockIdx.x * 64;
  const int h  = blockIdx.y, b = blockIdx.z;
  const float scale = 0.08838834764831845f;      // 1/sqrt(128)

  // Q fragments (pre-scaled), held in registers for the whole pass
  v16bf qf[4];
  {
    const float* qp = Q + ((size_t)(b * SEQ_ + q0 + wid * 16 + ln)) * HID_ + h * HD_;
#pragma unroll
    for (int c = 0; c < 4; ++c)
#pragma unroll
      for (int j = 0; j < 16; ++j) {
        int k = c * 32 + 8 * g + (j & 7) + ((j >> 3) << 4);
        qf[c][j] = f2bf16(qp[k] * scale);
      }
  }

  v8f o[8];
#pragma unroll
  for (int i = 0; i < 8; ++i) o[i] = {};
  float rm[8], rs[8];
#pragma unroll
  for (int r = 0; r < 8; ++r) { rm[r] = -1e30f; rs[r] = 0.f; }

  const unsigned ldsK = (unsigned)(uintptr_t)&sKf[0];
  const unsigned ldsV = (unsigned)(uintptr_t)&sVf[0];
  const int kb_end = q0 / 32 + 2;                // causal early exit

  for (int kb = 0; kb < kb_end; ++kb) {
    const int s0 = kb * 32;
    if (wid == 0) {                              // one TDM issue per block
      const float* kg = Kc + (((size_t)(b * SEQ_ + s0)) * NH_ + h) * HD_;
      const float* vg = Vc + (((size_t)(b * SEQ_ + s0)) * NH_ + h) * HD_;
      tdm_load_tile_32x128(ldsK, kg);
      tdm_load_tile_32x128(ldsV, vg);
      __builtin_amdgcn_s_wait_tensorcnt(0);
    }
    __syncthreads();

    // ---- scores: S = Qhat @ K^T  (two 16x16 key tiles) ----
    v8f s0a = {}; v8f s1a = {};
#pragma unroll
    for (int c = 0; c < 4; ++c) {
      v16bf kf0, kf1;
#pragma unroll
      for (int j = 0; j < 16; ++j) {
        int k = c * 32 + 8 * g + (j & 7) + ((j >> 3) << 4);
        kf0[j] = f2bf16(sKf[ln * LKV + k]);
        kf1[j] = f2bf16(sKf[(16 + ln) * LKV + k]);
      }
      s0a = WMMA_BF16(qf[c], kf0, s0a);
      s1a = WMMA_BF16(qf[c], kf1, s1a);
    }

    // ---- causal mask + online softmax (16-lane row reductions) ----
    unsigned short* pw = &sP[wid][0];
#pragma unroll
    for (int r = 0; r < 8; ++r) {
      int mrow = r + 8 * g;
      int qidx = q0 + wid * 16 + mrow;
      int kx0 = s0 + ln, kx1 = s0 + 16 + ln;
      float v0 = (kx0 <= qidx) ? s0a[r] : -1e30f;
      float v1 = (kx1 <= qidx) ? s1a[r] : -1e30f;
      float mx = fmaxf(v0, v1);
#pragma unroll
      for (int msk = 1; msk < 16; msk <<= 1) mx = fmaxf(mx, __shfl_xor(mx, msk, 32));
      float nm = fmaxf(rm[r], mx);
      float p0 = (v0 > -1e29f) ? __expf(v0 - nm) : 0.f;
      float p1 = (v1 > -1e29f) ? __expf(v1 - nm) : 0.f;
      float corr = __expf(rm[r] - nm);
      float ps = p0 + p1;
#pragma unroll
      for (int msk = 1; msk < 16; msk <<= 1) ps += __shfl_xor(ps, msk, 32);
      rs[r] = rs[r] * corr + ps;
      rm[r] = nm;
#pragma unroll
      for (int t2 = 0; t2 < 8; ++t2) o[t2][r] *= corr;
      pw[mrow * 34 + ln]      = f2bf(p0);
      pw[mrow * 34 + 16 + ln] = f2bf(p1);
    }

    // ---- O += P @ V ----
    v16bf pf;
#pragma unroll
    for (int j = 0; j < 16; ++j) {
      int k = 8 * g + (j & 7) + ((j >> 3) << 4);
      pf[j] = bfbits(pw[ln * 34 + k]);
    }
#pragma unroll
    for (int t2 = 0; t2 < 8; ++t2) {
      v16bf vf;
#pragma unroll
      for (int j = 0; j < 16; ++j) {
        int kk = 8 * g + (j & 7) + ((j >> 3) << 4);
        vf[j] = f2bf16(sVf[kk * LKV + t2 * 16 + ln]);
      }
      o[t2] = WMMA_BF16(pf, vf, o[t2]);
    }
    __syncthreads();
  }

  // ---- normalize + write attn(t, h*HD + d) ----
#pragma unroll
  for (int r = 0; r < 8; ++r) {
    float inv = 1.f / rs[r];
    int mr = q0 + wid * 16 + r + 8 * g;
    float* op = O + ((size_t)(b * SEQ_ + mr)) * HID_ + h * HD_;
#pragma unroll
    for (int t2 = 0; t2 < 8; ++t2)
      op[t2 * 16 + ln] = o[t2][r] * inv;
  }
}

// =====================================================================
extern "C" void kernel_launch(void* const* d_in, const int* in_sizes, int n_in,
                              void* d_out, int out_size, void* d_ws, size_t ws_size,
                              hipStream_t stream) {
  (void)in_sizes; (void)n_in; (void)out_size; (void)ws_size;
  const float* hidden  = (const float*)d_in[0];
  const float* cosb    = (const float*)d_in[1];
  const float* sinb    = (const float*)d_in[2];
  const float* w_qkv   = (const float*)d_in[3];
  const float* b_qkv   = (const float*)d_in[4];
  const float* w_dense = (const float*)d_in[5];
  const float* b_dense = (const float*)d_in[6];
  const float* k_in    = (const float*)d_in[7];
  const float* v_in    = (const float*)d_in[8];
  const int*   slots   = (const int*)d_in[9];

  float* out   = (float*)d_out;
  float* kcout = out + (size_t)T_ * HID_;
  float* vcout = kcout + (size_t)NSLOTS_ * NH_ * HD_;

  float* qkv  = (float*)d_ws;                       // T * 3*HID
  float* qbuf = qkv + (size_t)T_ * 3 * HID_;        // T * HID
  float* attn = qkv;                                // reuse after RoPE

  // 1) QKV projection
  gemm_bias_wmma<<<dim3(3 * HID_ / 64, T_ / 64), 256, 0, stream>>>(
      hidden, w_qkv, b_qkv, qkv, T_, 3 * HID_, HID_);

  // 2) initialize cache outputs from input caches
  size_t cbytes = (size_t)NSLOTS_ * NH_ * HD_ * sizeof(float);
  hipMemcpyAsync(kcout, k_in, cbytes, hipMemcpyDeviceToDevice, stream);
  hipMemcpyAsync(vcout, v_in, cbytes, hipMemcpyDeviceToDevice, stream);

  // 3) RoPE + scatter
  int total = T_ * NH_ * HD_;
  rope_scatter<<<(total + 255) / 256, 256, 0, stream>>>(
      qkv, cosb, sinb, slots, qbuf, kcout, vcout);

  // 4) causal flash attention
  flash_attn_wmma<<<dim3(SEQ_ / 64, NH_, BATCH_), 128, 0, stream>>>(
      qbuf, kcout, vcout, attn);

  // 5) output projection
  gemm_bias_wmma<<<dim3(HID_ / 64, T_ / 64), 256, 0, stream>>>(
      attn, w_dense, b_dense, out, T_, HID_, HID_);
}
